// BertTemporalOrdering_2164663517563
// MI455X (gfx1250) — matrix operand order repaired and checked by reading, not compiled, not run on hardware
//
#include <hip/hip_runtime.h>
#include <math.h>

// ---------------- problem constants (from reference) ----------------
constexpr int Bc = 16;
constexpr int Sc = 1024;
constexpr int Hc = 768;
constexpr int Jc = 5;
constexpr float NEGINF = -1e9f;

constexpr int ROWS = 64;          // s-rows per tile
constexpr int LDK  = Hc + 16;     // LDS row stride (elements), 16B-aligned rows, bank skew
constexpr int OUTROW = Jc * 2 * Hc + Hc;   // 8448
constexpr int OUT2_OFF = Bc * OUTROW;      // second output (full_emb) offset

constexpr int WN = Jc * Hc * Hc;           // attn_W elements  (2,949,120)
constexpr int HN = Bc * Sc * Hc;           // hidden elements  (12,582,912)

// ---------------- vector types ----------------
typedef __attribute__((ext_vector_type(16))) __bf16          v16bf;
typedef __attribute__((ext_vector_type(8)))  float           v8f;
typedef __attribute__((ext_vector_type(8)))  unsigned short  v8u16;
typedef __attribute__((ext_vector_type(16))) unsigned short  v16u16;
typedef __attribute__((ext_vector_type(4)))  unsigned short  u16x4;

__device__ __forceinline__ unsigned short f32_to_bf16(float f) {
    unsigned u = __builtin_bit_cast(unsigned, f);
    unsigned r = u + 0x7FFFu + ((u >> 16) & 1u);   // round-to-nearest-even
    return (unsigned short)(r >> 16);
}

// branchless tanh: clamp then (e^{2x}-1)/(e^{2x}+1); no libm EXEC divergence
__device__ __forceinline__ float tanh_fast(float x) {
    x = fminf(10.f, fmaxf(-10.f, x));
    float e = __expf(2.f * x);
    return (e - 1.f) * __builtin_amdgcn_rcpf(e + 1.f);
}

__device__ __forceinline__ v16bf load_frag16(const unsigned short* p) {
    v8u16 lo = *(const v8u16*)p;
    v8u16 hi = *(const v8u16*)(p + 16);
    v16u16 u = __builtin_shufflevector(lo, hi, 0,1,2,3,4,5,6,7,8,9,10,11,12,13,14,15);
    return __builtin_bit_cast(v16bf, u);
}

// ---------------- pre-pass: fp32 -> bf16 (vectorized, grid-stride) ----------------
__global__ __launch_bounds__(256)
void cvt_bf16_kernel(const float* __restrict__ src, unsigned short* __restrict__ dst, int n4) {
    for (int i = blockIdx.x * blockDim.x + threadIdx.x; i < n4; i += gridDim.x * blockDim.x) {
        const float4 v = *(const float4*)(src + (size_t)i * 4);
        u16x4 p;
        p[0] = f32_to_bf16(v.x); p[1] = f32_to_bf16(v.y);
        p[2] = f32_to_bf16(v.z); p[3] = f32_to_bf16(v.w);
        *(u16x4*)(dst + (size_t)i * 4) = p;
    }
}

// ---------------- main fused kernel: one block per (b, j) ----------------
__global__ __launch_bounds__(256, 1)
void span_pool_kernel(const float*          __restrict__ hidden,    // [B,S,H] fp32 (pools)
                      const unsigned short* __restrict__ hiddenB,   // [B,S,H] bf16 (GEMM A)
                      const float*          __restrict__ full_emb,  // [B,H]
                      const unsigned short* __restrict__ attn_Wb,   // [J,H,H] bf16 (GEMM B)
                      const float*          __restrict__ attn_b,    // [J,H]
                      const float*          __restrict__ attn_q,    // [J,H]
                      const int*            __restrict__ spans,     // [B,J,2]
                      float*                __restrict__ out)
{
    extern __shared__ unsigned char smem_raw[];
    unsigned short* As      = (unsigned short*)smem_raw;            // ROWS * LDK bf16
    float*          scoreAll= (float*)(smem_raw + ROWS * LDK * 2);  // Sc floats
    float*          sc      = scoreAll + Sc;                        // ROWS floats
    float*          red     = sc + ROWS;                            // 256 floats

    const int blk  = blockIdx.x;
    const int b    = blk / Jc;
    const int j    = blk % Jc;
    const int tid  = threadIdx.x;
    const int lane = tid & 31;
    const int wave = tid >> 5;
    const int nlane = lane & 15;

    const int start = spans[(b * Jc + j) * 2 + 0];
    const int end   = spans[(b * Jc + j) * 2 + 1];
    const int count = end - start;                 // spans sorted -> >= 0

    float macc[3] = {0.f, 0.f, 0.f};
    float attc[3] = {0.f, 0.f, 0.f};
    float invTot  = 0.f;

    if (count > 0) {
        const int khalf = (lane >= 16) ? 8 : 0;    // A-fragment K sub-offset per half-wave
        const int kb16  = (lane >= 16) ? 16 : 0;   // unused; B fragment also uses khalf

        for (int s0 = (start & ~(ROWS - 1)); s0 < end; s0 += ROWS) {
            __syncthreads();   // previous tile's LDS consumers done

            if (tid < ROWS) sc[tid] = 0.f;

            // ---- stage bf16 A tile into LDS: straight 16B copies ----
            for (int e = tid * 8; e < ROWS * Hc; e += 256 * 8) {
                int r = e / Hc, c = e % Hc;
                v8u16 v = *(const v8u16*)(hiddenB + ((size_t)b * Sc + (s0 + r)) * Hc + c);
                *(v8u16*)(As + r * LDK + c) = v;
            }

            // ---- mean-pool partial sums (fp32 source, column-partitioned) ----
            {
                int r0 = max(start, s0), r1 = min(end, s0 + ROWS);
                for (int s = r0; s < r1; ++s) {
                    const float* hp = hidden + ((size_t)b * Sc + s) * Hc;
                    macc[0] += hp[tid];
                    macc[1] += hp[tid + 256];
                    macc[2] += hp[tid + 512];
                }
            }
            __syncthreads();   // A tile visible

            // ---- projection GEMM fused with tanh + q-dot ----
            // N-blocked: each wave computes a 64x32 strip per pass (2 o-groups),
            // so each A fragment (1 ds_load_b128 pair) feeds TWO WMMAs.
            for (int pass = 0; pass < 3; ++pass) {
                const int o0 = pass * 256 + wave * 32 + nlane;
                const int o1 = o0 + 16;
                const unsigned short* wrow0 = attn_Wb + ((size_t)j * Hc + o0) * Hc;
                const unsigned short* wrow1 = attn_Wb + ((size_t)j * Hc + o1) * Hc;

                v8f acc0[4], acc1[4];
                const v8f z8 = {0.f, 0.f, 0.f, 0.f, 0.f, 0.f, 0.f, 0.f};
#pragma unroll
                for (int s = 0; s < 4; ++s) { acc0[s] = z8; acc1[s] = z8; }

                for (int k0 = 0; k0 < Hc; k0 += 32) {
                    const v16bf bfrag0 = load_frag16(wrow0 + k0 + khalf);
                    const v16bf bfrag1 = load_frag16(wrow1 + k0 + khalf);
                    if (k0 + 64 < Hc) {
                        __builtin_prefetch(wrow0 + k0 + 64, 0, 1);
                        __builtin_prefetch(wrow1 + k0 + 64, 0, 1);
                    }
#pragma unroll
                    for (int sub = 0; sub < 4; ++sub) {
                        const unsigned short* ap =
                            As + (size_t)(sub * 16 + nlane) * LDK + k0 + khalf;
                        v8u16 lo = *(const v8u16*)ap;          // K = k0+khalf    .. +8
                        v8u16 hi = *(const v8u16*)(ap + 16);   // K = k0+16+khalf .. +8
                        v16u16 au = __builtin_shufflevector(lo, hi,
                                        0,1,2,3,4,5,6,7,8,9,10,11,12,13,14,15);
                        const v16bf afrag = __builtin_bit_cast(v16bf, au);
                        acc0[sub] = __builtin_amdgcn_wmma_f32_16x16x32_bf16(
                                        false, afrag, false, bfrag0,
                                        (short)0, acc0[sub], false, false);
                        acc1[sub] = __builtin_amdgcn_wmma_f32_16x16x32_bf16(
                                        false, afrag, false, bfrag1,
                                        (short)0, acc1[sub], false, false);
                    }
                }

                // epilogue: bias + branchless tanh + q weight for both o-groups,
                // folded into a single half-wave reduction + one ds_add_f32
                const float bias0 = attn_b[(size_t)j * Hc + o0];
                const float qv0   = attn_q[(size_t)j * Hc + o0];
                const float bias1 = attn_b[(size_t)j * Hc + o1];
                const float qv1   = attn_q[(size_t)j * Hc + o1];
#pragma unroll
                for (int sub = 0; sub < 4; ++sub) {
#pragma unroll
                    for (int r = 0; r < 8; ++r) {
                        float v = qv0 * tanh_fast(acc0[sub][r] + bias0)
                                + qv1 * tanh_fast(acc1[sub][r] + bias1);
                        v += __shfl_xor(v, 1);
                        v += __shfl_xor(v, 2);
                        v += __shfl_xor(v, 4);
                        v += __shfl_xor(v, 8);
                        if (nlane == 0) {
                            int m = sub * 16 + r + ((lane >= 16) ? 8 : 0);
                            atomicAdd(&sc[m], v);   // ds_add_f32
                        }
                    }
                }
            }
            __syncthreads();
            if (tid < ROWS) {
                int s = s0 + tid;
                if (s < Sc) scoreAll[s] = sc[tid];
            }
        }
        __syncthreads();

        // ---- masked softmax over the span ----
        float m = NEGINF;
#pragma unroll
        for (int i = 0; i < 4; ++i) {
            int s = tid * 4 + i;
            if (s >= start && s < end) m = fmaxf(m, scoreAll[s]);
        }
        red[tid] = m;
        __syncthreads();
        for (int off = 128; off > 0; off >>= 1) {
            if (tid < off) red[tid] = fmaxf(red[tid], red[tid + off]);
            __syncthreads();
        }
        const float mx = red[0];
        __syncthreads();

        float ps = 0.f;
#pragma unroll
        for (int i = 0; i < 4; ++i) {
            int s = tid * 4 + i;
            float e = (s >= start && s < end) ? __expf(scoreAll[s] - mx) : 0.f;
            scoreAll[s] = e;     // thread-exclusive slots
            ps += e;
        }
        red[tid] = ps;
        __syncthreads();
        for (int off = 128; off > 0; off >>= 1) {
            if (tid < off) red[tid] += red[tid + off];
            __syncthreads();
        }
        invTot = 1.f / red[0];
        __syncthreads();

        // ---- attention pool (column-partitioned, fp32 source) ----
        for (int s = start; s < end; ++s) {
            const float a = scoreAll[s];
            const float* hp = hidden + ((size_t)b * Sc + s) * Hc;
            attc[0] += a * hp[tid];
            attc[1] += a * hp[tid + 256];
            attc[2] += a * hp[tid + 512];
        }
    }

    // ---- write outputs ----
    const float invc = (count > 0) ? (1.f / (float)count) : 0.f;
    float* frow = out + (size_t)b * OUTROW;
#pragma unroll
    for (int i = 0; i < 3; ++i) {
        const int c = tid + i * 256;
        frow[(size_t)j * 2 * Hc + c]      = macc[i] * invc;
        frow[(size_t)j * 2 * Hc + Hc + c] = attc[i] * invTot;
        if (j == 0) {
            const float fe = full_emb[(size_t)b * Hc + c];
            frow[(size_t)Jc * 2 * Hc + c] = fe;
            out[OUT2_OFF + (size_t)b * Hc + c] = fe;
        }
    }
}

extern "C" void kernel_launch(void* const* d_in, const int* in_sizes, int n_in,
                              void* d_out, int out_size, void* d_ws, size_t ws_size,
                              hipStream_t stream) {
    (void)in_sizes; (void)n_in; (void)ws_size; (void)out_size;
    const float* hidden   = (const float*)d_in[0];
    const float* full_emb = (const float*)d_in[1];
    const float* attn_W   = (const float*)d_in[2];
    const float* attn_b   = (const float*)d_in[3];
    const float* attn_q   = (const float*)d_in[4];
    const int*   spans    = (const int*)  d_in[5];
    float* out = (float*)d_out;

    // workspace layout: bf16 W, then bf16 hidden (both 16B aligned)
    unsigned short* wsW = (unsigned short*)d_ws;
    unsigned short* wsH = wsW + WN;          // WN*2 bytes = 5,898,240 (16B multiple)

    cvt_bf16_kernel<<<dim3(1024), dim3(256), 0, stream>>>(attn_W, wsW, WN / 4);
    cvt_bf16_kernel<<<dim3(2048), dim3(256), 0, stream>>>(hidden, wsH, HN / 4);

    constexpr size_t SMEM = (size_t)ROWS * LDK * 2      // bf16 A tile
                          + (size_t)Sc * 4              // scores / alphas
                          + (size_t)ROWS * 4            // per-tile score accum
                          + 256 * 4;                    // reduction scratch

    span_pool_kernel<<<dim3(Bc * Jc), dim3(256), SMEM, stream>>>(
        hidden, wsH, full_emb, wsW, attn_b, attn_q, spans, out);
}
